// LatentDiffModel_33535104647912
// MI455X (gfx1250) — matrix operand order repaired and checked by reading, not compiled, not run on hardware
//
#include <hip/hip_runtime.h>

// Problem constants (fixed by the reference).
#define B_  8
#define L_  1024
#define D_  512
#define K_  4096
#define M_  (B_ * L_)          // 8192 rows
#define PAD_NUM_   2049
#define PAD_CHAIN_ 63
#define PAD_ENT_   63

typedef _Float16 v16h __attribute__((ext_vector_type(16)));
typedef _Float16 v4h  __attribute__((ext_vector_type(4)));
typedef float    v8f  __attribute__((ext_vector_type(8)));

// ---------------------------------------------------------------------------
// Stage A: pred = input_hidden + pos[sidx] + chain[cidx] + ent[eidx]
//          split fp32 -> (hi, lo) fp16 pair, and x2[m] = ||pred_m||^2.
// ---------------------------------------------------------------------------
__global__ __launch_bounds__(128) void ldm_prep_rows(
    const float* __restrict__ ih, const float* __restrict__ pos,
    const float* __restrict__ ch, const float* __restrict__ en,
    const unsigned char* __restrict__ mask,
    const int* __restrict__ res, const int* __restrict__ cix,
    const int* __restrict__ eix,
    _Float16* __restrict__ pH, _Float16* __restrict__ pL,
    float* __restrict__ x2)
{
  const int i = blockIdx.x;               // row in [0, 8192)
  const bool pad = (mask[i] == 0);
  const int s = pad ? PAD_NUM_   : res[i];
  const int c = pad ? PAD_CHAIN_ : cix[i];
  const int e = pad ? PAD_ENT_   : eix[i];

  const int d0 = threadIdx.x * 4;
  const float4 a = *(const float4*)(ih  + (size_t)i * D_ + d0);
  const float4 p = *(const float4*)(pos + (size_t)s * D_ + d0);
  const float4 q = *(const float4*)(ch  + (size_t)c * D_ + d0);
  const float4 r = *(const float4*)(en  + (size_t)e * D_ + d0);

  float v[4] = { a.x + p.x + q.x + r.x, a.y + p.y + q.y + r.y,
                 a.z + p.z + q.z + r.z, a.w + p.w + q.w + r.w };
  v4h hi, lo;
  float ss = 0.f;
#pragma unroll
  for (int t = 0; t < 4; t++) {
    _Float16 h = (_Float16)v[t];          // RTE
    hi[t] = h;
    lo[t] = (_Float16)(v[t] - (float)h);  // residual
    ss += v[t] * v[t];
  }
  *(v4h*)(pH + (size_t)i * D_ + d0) = hi;
  *(v4h*)(pL + (size_t)i * D_ + d0) = lo;

  __shared__ float red[128];
  red[threadIdx.x] = ss;
  __syncthreads();
#pragma unroll
  for (int sdt = 64; sdt > 0; sdt >>= 1) {
    if (threadIdx.x < sdt) red[threadIdx.x] += red[threadIdx.x + sdt];
    __syncthreads();
  }
  if (threadIdx.x == 0) x2[i] = red[0];
}

// ---------------------------------------------------------------------------
// Stage B: codebook split (hi, lo) fp16 and c2[k] = ||c_k||^2.
// ---------------------------------------------------------------------------
__global__ __launch_bounds__(128) void ldm_prep_codebook(
    const float* __restrict__ cb,
    _Float16* __restrict__ cH, _Float16* __restrict__ cL,
    float* __restrict__ c2)
{
  const int k = blockIdx.x;               // row in [0, 4096)
  const int d0 = threadIdx.x * 4;
  const float4 a = *(const float4*)(cb + (size_t)k * D_ + d0);
  float v[4] = { a.x, a.y, a.z, a.w };
  v4h hi, lo;
  float ss = 0.f;
#pragma unroll
  for (int t = 0; t < 4; t++) {
    _Float16 h = (_Float16)v[t];
    hi[t] = h;
    lo[t] = (_Float16)(v[t] - (float)h);
    ss += v[t] * v[t];
  }
  *(v4h*)(cH + (size_t)k * D_ + d0) = hi;
  *(v4h*)(cL + (size_t)k * D_ + d0) = lo;

  __shared__ float red[128];
  red[threadIdx.x] = ss;
  __syncthreads();
#pragma unroll
  for (int sdt = 64; sdt > 0; sdt >>= 1) {
    if (threadIdx.x < sdt) red[threadIdx.x] += red[threadIdx.x + sdt];
    __syncthreads();
  }
  if (threadIdx.x == 0) c2[k] = red[0];
}

// ---------------------------------------------------------------------------
// Stage C: out[m,n] = x2[m] + c2[n] - 2 * (xh.ch + xh.cl + xl.ch)
// 128x128 tile per block; 8 wave32s, each a 32x64 sub-tile (2x4 WMMA frags).
// Double-buffered LDS (2 x 4 tiles x 128 rows x 40-half pitch = 80 KB),
// filled with ASYNCcnt-tracked global_load_async_to_lds_b128.
// ---------------------------------------------------------------------------
#define BM_    128
#define BN_    128
#define BK_    32
#define PITCH_ 40     // halfs per LDS row (80 bytes: 16B-aligned, bank-spread)
#define TILEH_ (BM_ * PITCH_)   // halfs per tile per buffer

#if __has_builtin(__builtin_amdgcn_s_wait_asynccnt)
#define WAIT_ASYNC(n) __builtin_amdgcn_s_wait_asynccnt((unsigned short)(n))
#else
#define WAIT_ASYNC(n) asm volatile("s_wait_asynccnt %0" :: "n"(n) : "memory")
#endif

// LDS byte address = low 32 bits of the flat pointer (aperture maps
// addr[31:0] -> LDS offset; CDNA5 ISA 10.2).
static __device__ inline unsigned lds_addr(const void* p) {
  return (unsigned)(size_t)p;
}

// ASYNCcnt-tracked direct global->LDS copy, 16 bytes per lane.
static __device__ inline void async_ldst_b128(unsigned lds, const void* gaddr) {
  asm volatile("global_load_async_to_lds_b128 %0, %1, off"
               :: "v"(lds), "v"(gaddr)
               : "memory");
}

// A-fragment (16x32 f16, ISA 7.12.2): lane row = lane&15;
// lanes 0-15 hold K {0..7, 16..23}; lanes 16-31 hold K {8..15, 24..31}.
static __device__ inline v16h frag_a(const _Float16* t, int row, int grp) {
  const uint4* p0 = (const uint4*)(t + row * PITCH_ + grp * 8);
  const uint4* p1 = (const uint4*)(t + row * PITCH_ + 16 + grp * 8);
  union { uint4 u[2]; v16h v; } r;
  r.u[0] = *p0; r.u[1] = *p1;
  return r.v;
}

// B-fragment (32x16 f16): lane col = lane&15;
// lanes 0-15 hold K {0..7, 16..23}; lanes 16-31 hold K {8..15, 24..31}.
static __device__ inline v16h frag_b(const _Float16* t, int row, int grp) {
  const uint4* p0 = (const uint4*)(t + row * PITCH_ + grp * 8);
  const uint4* p1 = (const uint4*)(t + row * PITCH_ + 16 + grp * 8);
  union { uint4 u[2]; v16h v; } r;
  r.u[0] = *p0; r.u[1] = *p1;
  return r.v;
}

__global__ __launch_bounds__(256) void ldm_l2_gemm(
    const _Float16* __restrict__ pH, const _Float16* __restrict__ pL,
    const _Float16* __restrict__ cH, const _Float16* __restrict__ cL,
    const float* __restrict__ x2, const float* __restrict__ c2,
    float* __restrict__ out)
{
  __shared__ __attribute__((aligned(16))) _Float16 sAh[2 * TILEH_];
  __shared__ __attribute__((aligned(16))) _Float16 sAl[2 * TILEH_];
  __shared__ __attribute__((aligned(16))) _Float16 sBh[2 * TILEH_];
  __shared__ __attribute__((aligned(16))) _Float16 sBl[2 * TILEH_];

  const int tid  = threadIdx.x;
  const int m0   = blockIdx.y * BM_;
  const int n0   = blockIdx.x * BN_;
  const int wid  = tid >> 5;
  const int lane = tid & 31;
  const int laneR = lane & 15;
  const int grp   = lane >> 4;          // 0 or 1 (lane half-group)
  const int mOff  = (wid & 3) * 32;     // wave sub-tile: 32 rows
  const int nOff  = (wid >> 2) * 64;    //                64 cols

  v8f acc[2][4] = {};

  // Cooperative tile fill: 16B chunk per (thread, h) slot; 8 async b128/thread.
  const int qr = tid >> 2;              // 0..63 (row; second row = +64)
  const int qc = (tid & 3) * 8;         // half offset inside 32-half row

  // Issue the 8 async global->LDS loads for k-chunk `kc` into buffer `buf`.
  auto issue = [&](int kc, int buf) {
    const int bo = buf * TILEH_;
#pragma unroll
    for (int h = 0; h < 2; h++) {
      const int r = qr + h * 64;
      const int ldst = bo + r * PITCH_ + qc;
      async_ldst_b128(lds_addr(sAh + ldst), pH + (size_t)(m0 + r) * D_ + kc + qc);
      async_ldst_b128(lds_addr(sAl + ldst), pL + (size_t)(m0 + r) * D_ + kc + qc);
      async_ldst_b128(lds_addr(sBh + ldst), cH + (size_t)(n0 + r) * D_ + kc + qc);
      async_ldst_b128(lds_addr(sBl + ldst), cL + (size_t)(n0 + r) * D_ + kc + qc);
    }
  };

  issue(0, 0);                          // prologue fill of buffer 0

  const int NK = D_ / BK_;              // 16 k-steps
  for (int ks = 0; ks < NK; ks++) {
    const int p  = ks & 1;
    const int bo = p * TILEH_;

    if (ks + 1 < NK) {
      issue((ks + 1) * BK_, p ^ 1);     // prefetch next chunk into other buffer
      WAIT_ASYNC(8);                    // in-order: batch `ks` (8 loads) retired
    } else {
      WAIT_ASYNC(0);
    }
    __syncthreads();                    // all waves' LDS writes visible

    v16h ah[2], al[2];
#pragma unroll
    for (int mf = 0; mf < 2; mf++) {
      ah[mf] = frag_a(sAh + bo, mOff + mf * 16 + laneR, grp);
      al[mf] = frag_a(sAl + bo, mOff + mf * 16 + laneR, grp);
    }
#pragma unroll
    for (int nf = 0; nf < 4; nf++) {
      const v16h bh = frag_b(sBh + bo, nOff + nf * 16 + laneR, grp);
      const v16h bl = frag_b(sBl + bo, nOff + nf * 16 + laneR, grp);
#pragma unroll
      for (int mf = 0; mf < 2; mf++) {
        acc[mf][nf] = __builtin_amdgcn_wmma_f32_16x16x32_f16(
            false, ah[mf], false, bh, (short)0, acc[mf][nf], false, false);
        acc[mf][nf] = __builtin_amdgcn_wmma_f32_16x16x32_f16(
            false, ah[mf], false, bl, (short)0, acc[mf][nf], false, false);
        acc[mf][nf] = __builtin_amdgcn_wmma_f32_16x16x32_f16(
            false, al[mf], false, bh, (short)0, acc[mf][nf], false, false);
      }
    }
    __syncthreads();                    // buffer p free for overwrite at ks+2
  }

  // Epilogue: D VGPR j holds (M = base + j + grp*8, N = n base + laneR).
  // Non-temporal stores: keep the 134 MB output stream from evicting the
  // L2-resident pred/codebook operands (24 MB << 192 MB L2).
#pragma unroll
  for (int nf = 0; nf < 4; nf++) {
    const int n = n0 + nOff + nf * 16 + laneR;
    const float c2n = c2[n];
#pragma unroll
    for (int mf = 0; mf < 2; mf++) {
      const int mb = m0 + mOff + mf * 16 + grp * 8;
#pragma unroll
      for (int j = 0; j < 8; j++) {
        const int m = mb + j;
        __builtin_nontemporal_store(x2[m] + c2n - 2.0f * acc[mf][nf][j],
                                    &out[(size_t)m * K_ + n]);
      }
    }
  }
}

// ---------------------------------------------------------------------------
// Host-side launch (graph-capture safe: stream-ordered, no alloc/sync).
// Workspace layout (bytes):
//   predH : [0,        8388608)   8192*512 f16
//   predL : [8388608, 16777216)
//   cbH   : [16777216,20971520)   4096*512 f16
//   cbL   : [20971520,25165824)
//   x2    : [25165824,25198592)   8192 f32
//   c2    : [25198592,25214976)   4096 f32
// ---------------------------------------------------------------------------
extern "C" void kernel_launch(void* const* d_in, const int* in_sizes, int n_in,
                              void* d_out, int out_size, void* d_ws, size_t ws_size,
                              hipStream_t stream) {
  const float* input_hidden = (const float*)d_in[0];
  const float* pos_table    = (const float*)d_in[1];
  const float* chain_emb    = (const float*)d_in[2];
  const float* entity_emb   = (const float*)d_in[3];
  const float* codebook     = (const float*)d_in[4];
  const unsigned char* mask = (const unsigned char*)d_in[5];  // jnp.bool_: 1B
  const int* res_rel        = (const int*)d_in[6];
  const int* chain_idx      = (const int*)d_in[7];
  const int* entity_idx     = (const int*)d_in[8];
  float* out = (float*)d_out;

  char* w = (char*)d_ws;
  _Float16* predH = (_Float16*)(w);
  _Float16* predL = (_Float16*)(w + 8388608);
  _Float16* cbH   = (_Float16*)(w + 16777216);
  _Float16* cbL   = (_Float16*)(w + 20971520);
  float*    x2    = (float*)(w + 25165824);
  float*    c2    = (float*)(w + 25198592);

  ldm_prep_rows<<<M_, 128, 0, stream>>>(
      input_hidden, pos_table, chain_emb, entity_emb,
      mask, res_rel, chain_idx, entity_idx, predH, predL, x2);

  ldm_prep_codebook<<<K_, 128, 0, stream>>>(codebook, cbH, cbL, c2);

  dim3 grid(K_ / BN_, M_ / BM_);   // (32, 64)
  ldm_l2_gemm<<<grid, 256, 0, stream>>>(predH, predL, cbH, cbL, x2, c2, out);
}
